// RiftMambaModel_76553497084125
// MI455X (gfx1250) — matrix operand order repaired and checked by reading, not compiled, not run on hardware
//
#include <hip/hip_runtime.h>
#include <stdint.h>

// ---------------------------------------------------------------------------
// Problem constants (match reference)
// ---------------------------------------------------------------------------
constexpr int    Bb     = 4;
constexpr int    Ss     = 4096;
constexpr int    Dd     = 1024;
constexpr int    INNERc = 2048;      // D * EXP
constexpr int    KCONV  = 3;
constexpr size_t Mrows  = (size_t)Bb * Ss;   // 16384
constexpr float  LN_EPS = 1e-5f;

// ---------------------------------------------------------------------------
// Types for WMMA / async copy
// ---------------------------------------------------------------------------
typedef __bf16 bf16x16 __attribute__((ext_vector_type(16)));
typedef float  f32x8   __attribute__((ext_vector_type(8)));
typedef int    i32x4v  __attribute__((vector_size(16)));   // matches builtin param type

#define AS1 __attribute__((address_space(1)))
#define AS3 __attribute__((address_space(3)))

union FragBF {
    uint4   q[2];
    bf16x16 v;
};

__device__ __forceinline__ __bf16 f2bf(float f) {
    union { float f; uint32_t u; } v; v.f = f;
    uint32_t r = v.u + 0x7FFFu + ((v.u >> 16) & 1u);   // round-to-nearest-even
    unsigned short h = (unsigned short)(r >> 16);
    return __builtin_bit_cast(__bf16, h);
}
__device__ __forceinline__ float sigmoidf_(float x) { return 1.0f / (1.0f + __expf(-x)); }

// Async global->LDS 16-byte copy (CDNA5 GLOBAL_LOAD_ASYNC_TO_LDS_B128, ASYNCcnt)
__device__ __forceinline__ void async_ld_b128(const void* gsrc, void* ldst) {
    __builtin_amdgcn_global_load_async_to_lds_b128(
        (AS1 i32x4v*)gsrc, (AS3 i32x4v*)ldst, 0, 0);
}
__device__ __forceinline__ void wait_async0() {
    asm volatile("s_wait_asynccnt 0x0" ::: "memory");
}

// ---------------------------------------------------------------------------
// fp32 -> bf16 conversion (weights, etc.)
// ---------------------------------------------------------------------------
__global__ void cvt_bf16_kernel(const float* __restrict__ src, __bf16* __restrict__ dst, size_t n) {
    size_t i      = (size_t)blockIdx.x * blockDim.x + threadIdx.x;
    size_t stride = (size_t)gridDim.x * blockDim.x;
    for (; i < n; i += stride) dst[i] = f2bf(src[i]);
}

// ---------------------------------------------------------------------------
// LayerNorm over D per row, emit bf16 h
// ---------------------------------------------------------------------------
__global__ __launch_bounds__(256)
void ln_kernel(const float* __restrict__ x, const float* __restrict__ g,
               const float* __restrict__ b, __bf16* __restrict__ out) {
    const int row = blockIdx.x;
    const float* xr = x + (size_t)row * Dd;
    float s = 0.f, ss = 0.f;
    for (int i = threadIdx.x; i < Dd; i += 256) { float v = xr[i]; s += v; ss += v * v; }
#pragma unroll
    for (int o = 16; o > 0; o >>= 1) { s += __shfl_down(s, o, 32); ss += __shfl_down(ss, o, 32); }
    __shared__ float rs[8], rss[8];
    const int wv = threadIdx.x >> 5, ln = threadIdx.x & 31;
    if (ln == 0) { rs[wv] = s; rss[wv] = ss; }
    __syncthreads();
    if (threadIdx.x == 0) {
        float a = 0.f, c = 0.f;
#pragma unroll
        for (int i = 0; i < 8; i++) { a += rs[i]; c += rss[i]; }
        rs[0] = a; rss[0] = c;
    }
    __syncthreads();
    const float mean = rs[0] * (1.0f / Dd);
    const float var  = rss[0] * (1.0f / Dd) - mean * mean;
    const float rstd = rsqrtf(var + LN_EPS);
    for (int i = threadIdx.x; i < Dd; i += 256) {
        float h = (xr[i] - mean) * rstd * g[i] + b[i];
        out[(size_t)row * Dd + i] = f2bf(h);
    }
}

// ---------------------------------------------------------------------------
// Tiled bf16 WMMA GEMM:  out[M,N] = epilogue(A[M,K] * W[N,K]^T + bias[N])
//   Workgroup tile 128(M)x256(N), 8 waves, each wave = 64x64 (4x4 fragments,
//   128 f32 acc VGPRs), Kblock=32.  Double-buffered LDS filled by
//   GLOBAL_LOAD_ASYNC_TO_LDS_B128 (ASYNCcnt; no VGPR round-trip).
//   XOR-swizzled LDS: 16B chunk c of row r lands at chunk (c ^ (r & 3)),
//   making the ds_load_b128 fragment reads bank-conflict-free.
// ---------------------------------------------------------------------------
enum GemmMode { MODE_PJ = 0, MODE_SIG = 1, MODE_TANH = 2, MODE_OUT = 3 };

template <int MODE>
__global__ __launch_bounds__(256)
void wmma_gemm_kernel(const __bf16* __restrict__ A, const __bf16* __restrict__ W,
                      const float* __restrict__ bias, float* __restrict__ out,
                      int Ntot, int Kd,
                      const unsigned char* __restrict__ mask,
                      const float* __restrict__ resid) {
    constexpr int TILE_M = 128, TILE_N = 256, KBLK = 32;
    __shared__ __align__(16) __bf16 sA[2][TILE_M * KBLK];   // 2 x 8 KB
    __shared__ __align__(16) __bf16 sB[2][TILE_N * KBLK];   // 2 x 16 KB

    const int tid   = threadIdx.x;
    const int lane  = tid & 31;
    const int wave  = tid >> 5;
    const int wm    = wave & 1;       // 0..1 -> M half (64 rows)
    const int wn    = wave >> 1;      // 0..3 -> N quarter (64 cols)
    const int lhalf = lane >> 4;      // 0/1 within wave
    const int lm    = lane & 15;

    const size_t mBase = (size_t)blockIdx.y * TILE_M;
    const size_t nBase = (size_t)blockIdx.x * TILE_N;

    const f32x8 zf = {0.f, 0.f, 0.f, 0.f, 0.f, 0.f, 0.f, 0.f};
    f32x8 acc[4][4];
#pragma unroll
    for (int mi = 0; mi < 4; mi++)
#pragma unroll
        for (int ni = 0; ni < 4; ni++) acc[mi][ni] = zf;

    // cooperative async-load mapping (per K-tile):
    //   A: 128 rows x 4 chunks (16B) -> thread t covers row t>>1, chunks c0,c0+1
    //   B: 256 rows x 4 chunks       -> thread t covers row t, chunks 0..3
    const int ldRowA = tid >> 1;
    const int cA0    = (tid & 1) * 2;
    const int swA0   = cA0 ^ (ldRowA & 3);
    const int swA1   = (cA0 + 1) ^ (ldRowA & 3);
    const int swB    = tid & 3;

    const __bf16* gArow = A + (mBase + ldRowA) * (size_t)Kd + cA0 * 8;
    const __bf16* gWrow = W + (nBase + tid) * (size_t)Kd;

    const int nk = Kd / KBLK;

    auto loadTile = [&](int buf, int kb) {
        uint4* a4 = reinterpret_cast<uint4*>(sA[buf]) + ldRowA * 4;
        uint4* b4 = reinterpret_cast<uint4*>(sB[buf]) + tid * 4;
        async_ld_b128(gArow + kb,      a4 + swA0);
        async_ld_b128(gArow + kb + 8,  a4 + swA1);
        async_ld_b128(gWrow + kb,      b4 + (0 ^ swB));
        async_ld_b128(gWrow + kb + 8,  b4 + (1 ^ swB));
        async_ld_b128(gWrow + kb + 16, b4 + (2 ^ swB));
        async_ld_b128(gWrow + kb + 24, b4 + (3 ^ swB));
    };

    loadTile(0, 0);

    for (int ik = 0; ik < nk; ik++) {
        const int cur = ik & 1;
        wait_async0();
        __syncthreads();
        if (ik + 1 < nk) loadTile(cur ^ 1, (ik + 1) * KBLK);

        const uint4* sA4 = reinterpret_cast<const uint4*>(sA[cur]);
        const uint4* sB4 = reinterpret_cast<const uint4*>(sB[cur]);

        FragBF af[4], bfb[4];
#pragma unroll
        for (int ni = 0; ni < 4; ni++) {
            const int brow = wn * 64 + ni * 16 + lm;    // W row (== output col) in tile
            const int bsw  = brow & 3;
            // lane<16: K 0..15 ; lane>=16: K 16..31 (chunks 2*lhalf, 2*lhalf+1)
            bfb[ni].q[0] = sB4[brow * 4 + ((2 * lhalf) ^ bsw)];
            bfb[ni].q[1] = sB4[brow * 4 + ((2 * lhalf + 1) ^ bsw)];
        }
#pragma unroll
        for (int mi = 0; mi < 4; mi++) {
            const int row = wm * 64 + mi * 16 + lm;     // A row in tile
            const int asw = row & 3;
            // lane<16: K 0..15 ; lane>=16: K 16..31 (chunks 2*lhalf, 2*lhalf+1)
            af[mi].q[0] = sA4[row * 4 + ((2 * lhalf) ^ asw)];
            af[mi].q[1] = sA4[row * 4 + ((2 * lhalf + 1) ^ asw)];
        }
#pragma unroll
        for (int mi = 0; mi < 4; mi++)
#pragma unroll
            for (int ni = 0; ni < 4; ni++)
                acc[mi][ni] = __builtin_amdgcn_wmma_f32_16x16x32_bf16(
                    false, af[mi].v, false, bfb[ni].v, (short)0, acc[mi][ni], false, false);
        __syncthreads();
    }

    // Epilogue: C/D layout — lanes 0-15: N=lane, M=r ; lanes 16-31: N=lane-16, M=r+8
#pragma unroll
    for (int mi = 0; mi < 4; mi++) {
        const size_t m0 = mBase + wm * 64 + mi * 16 + lhalf * 8;
#pragma unroll
        for (int ni = 0; ni < 4; ni++) {
            const size_t n = nBase + wn * 64 + ni * 16 + lm;
            const float bia = bias[n];
#pragma unroll
            for (int r = 0; r < 8; r++) {
                const size_t m = m0 + r;
                const float  v = acc[mi][ni][r] + bia;
                const size_t o = m * (size_t)Ntot + n;
                if constexpr (MODE == MODE_PJ)       out[o] = mask[m] ? v : 0.f;
                else if constexpr (MODE == MODE_SIG) out[o] = sigmoidf_(v);
                else if constexpr (MODE == MODE_TANH)out[o] = tanhf(v);
                else                                 out[o] = v + resid[o];
            }
        }
    }
}

// ---------------------------------------------------------------------------
// Depthwise causal conv (K=3) + SiLU * sigmoid(gate) -> u (fp32 + bf16)
//   pj layout: [M, 2*INNER], proj = cols [0,INNER), gate = cols [INNER,2*INNER)
// ---------------------------------------------------------------------------
__global__ __launch_bounds__(256)
void conv_gate_kernel(const float* __restrict__ pj, const float* __restrict__ wconv,
                      const float* __restrict__ bconv, float* __restrict__ u,
                      __bf16* __restrict__ ubf) {
    const size_t total  = Mrows * (size_t)INNERc;
    size_t idx    = (size_t)blockIdx.x * blockDim.x + threadIdx.x;
    const size_t stride = (size_t)gridDim.x * blockDim.x;
    for (; idx < total; idx += stride) {
        const int    ch = (int)(idx % INNERc);
        const size_t ms = idx / INNERc;       // = b*S + s
        const int    s  = (int)(ms % Ss);
        const size_t rowOff = ms * (size_t)(2 * INNERc);
        const float w0 = wconv[ch * KCONV + 0];
        const float w1 = wconv[ch * KCONV + 1];
        const float w2 = wconv[ch * KCONV + 2];
        float c = bconv[ch];
        c += pj[rowOff + ch] * w2;                                              // proj[s]
        if (s >= 1) c += pj[rowOff - (size_t)(2 * INNERc)     + ch] * w1;       // proj[s-1]
        if (s >= 2) c += pj[rowOff - (size_t)(4 * INNERc)     + ch] * w0;       // proj[s-2]
        const float gate = pj[rowOff + INNERc + ch];
        const float uu = c * sigmoidf_(c) * sigmoidf_(gate);
        u[idx]   = uu;
        ubf[idx] = f2bf(uu);
    }
}

// ---------------------------------------------------------------------------
// Sequential selective-scan: one thread per (batch, channel)
// ---------------------------------------------------------------------------
__global__ __launch_bounds__(256)
void scan_kernel(const float* __restrict__ aA, const float* __restrict__ bA,
                 const float* __restrict__ cA, const float* __restrict__ uA,
                 const unsigned char* __restrict__ mask, __bf16* __restrict__ ybf) {
    const int gid = blockIdx.x * blockDim.x + threadIdx.x;   // [0, B*INNER)
    if (gid >= Bb * INNERc) return;
    const int ch = gid % INNERc;
    const int bb = gid / INNERc;
    const size_t base = (size_t)bb * Ss * INNERc + ch;
    const unsigned char* mrow = mask + (size_t)bb * Ss;
    float st = 0.f;
    for (int t = 0; t < Ss; t++) {
        const size_t i = base + (size_t)t * INNERc;
        const float at = aA[i], bt = bA[i], ct = cA[i], ut = uA[i];
        const float mt = mrow[t] ? 1.f : 0.f;
        st = at * st + bt * ut;
        const float y = (ct * st + ut) * mt;
        st *= mt;
        ybf[i] = f2bf(y);
    }
}

// ---------------------------------------------------------------------------
// Workspace layout (bytes, all naturally aligned; total ~708 MB)
// ---------------------------------------------------------------------------
constexpr size_t SZ_WIN  = (size_t)2 * INNERc * Dd * 2;         // 8 MB  bf16 w_in
constexpr size_t SZ_WSQ  = (size_t)INNERc * INNERc * 2;         // 8 MB  bf16 square weight
constexpr size_t SZ_WOUT = (size_t)Dd * INNERc * 2;             // 4 MB
constexpr size_t SZ_HB   = Mrows * (size_t)Dd * 2;              // 32 MB
constexpr size_t SZ_PJ   = Mrows * (size_t)(2 * INNERc) * 4;    // 256 MB
constexpr size_t SZ_U    = Mrows * (size_t)INNERc * 4;          // 128 MB
constexpr size_t SZ_UBF  = Mrows * (size_t)INNERc * 2;          // 64 MB

constexpr size_t OFF_WIN  = 0;
constexpr size_t OFF_WA   = OFF_WIN + SZ_WIN;
constexpr size_t OFF_WB   = OFF_WA + SZ_WSQ;
constexpr size_t OFF_WC   = OFF_WB + SZ_WSQ;
constexpr size_t OFF_WOUT = OFF_WC + SZ_WSQ;
constexpr size_t OFF_HB   = OFF_WOUT + SZ_WOUT;
constexpr size_t OFF_PJ   = OFF_HB + SZ_HB;
constexpr size_t OFF_U    = OFF_PJ + SZ_PJ;
constexpr size_t OFF_UBF  = OFF_U + SZ_U;
constexpr size_t OFF_C    = OFF_UBF + SZ_UBF;                   // tanh branch (128 MB)
constexpr size_t OFF_YBF  = OFF_C + SZ_U;                       // 64 MB
// a/b activations reuse the pj region (pj fully consumed by conv kernel):
constexpr size_t OFF_AACT = OFF_PJ;
constexpr size_t OFF_BACT = OFF_PJ + SZ_U;

extern "C" void kernel_launch(void* const* d_in, const int* in_sizes, int n_in,
                              void* d_out, int out_size, void* d_ws, size_t ws_size,
                              hipStream_t stream) {
    const float* x      = (const float*)d_in[0];
    const unsigned char* mask = (const unsigned char*)d_in[1];  // jnp bool -> 1 byte
    const float* ln_g   = (const float*)d_in[2];
    const float* ln_b   = (const float*)d_in[3];
    const float* w_in   = (const float*)d_in[4];
    const float* b_in   = (const float*)d_in[5];
    const float* w_conv = (const float*)d_in[6];
    const float* b_conv = (const float*)d_in[7];
    const float* w_a    = (const float*)d_in[8];
    const float* b_a    = (const float*)d_in[9];
    const float* w_b    = (const float*)d_in[10];
    const float* b_b    = (const float*)d_in[11];
    const float* w_c    = (const float*)d_in[12];
    const float* b_c    = (const float*)d_in[13];
    const float* w_out  = (const float*)d_in[14];
    const float* b_out  = (const float*)d_in[15];
    float* outp = (float*)d_out;

    char* ws = (char*)d_ws;
    __bf16* wInB  = (__bf16*)(ws + OFF_WIN);
    __bf16* wAB   = (__bf16*)(ws + OFF_WA);
    __bf16* wBB   = (__bf16*)(ws + OFF_WB);
    __bf16* wCB   = (__bf16*)(ws + OFF_WC);
    __bf16* wOutB = (__bf16*)(ws + OFF_WOUT);
    __bf16* hb    = (__bf16*)(ws + OFF_HB);
    float*  pj    = (float*)(ws + OFF_PJ);
    float*  uf    = (float*)(ws + OFF_U);
    __bf16* ubf   = (__bf16*)(ws + OFF_UBF);
    float*  aAct  = (float*)(ws + OFF_AACT);
    float*  bAct  = (float*)(ws + OFF_BACT);
    float*  cAct  = (float*)(ws + OFF_C);
    __bf16* ybf   = (__bf16*)(ws + OFF_YBF);

    // 1) weights -> bf16
    cvt_bf16_kernel<<<2048, 256, 0, stream>>>(w_in,  wInB,  (size_t)2 * INNERc * Dd);
    cvt_bf16_kernel<<<2048, 256, 0, stream>>>(w_a,   wAB,   (size_t)INNERc * INNERc);
    cvt_bf16_kernel<<<2048, 256, 0, stream>>>(w_b,   wBB,   (size_t)INNERc * INNERc);
    cvt_bf16_kernel<<<2048, 256, 0, stream>>>(w_c,   wCB,   (size_t)INNERc * INNERc);
    cvt_bf16_kernel<<<2048, 256, 0, stream>>>(w_out, wOutB, (size_t)Dd * INNERc);

    // 2) LayerNorm -> bf16 h
    ln_kernel<<<(int)Mrows, 256, 0, stream>>>(x, ln_g, ln_b, hb);

    // 3) pj = mask * (h @ w_in^T + b_in)    [M x 4096, fp32]
    {
        dim3 grid((2 * INNERc) / 256, (int)(Mrows / 128));
        wmma_gemm_kernel<MODE_PJ><<<grid, 256, 0, stream>>>(hb, wInB, b_in, pj,
                                                            2 * INNERc, Dd, mask, nullptr);
    }

    // 4) causal conv + gating -> u
    conv_gate_kernel<<<65536, 256, 0, stream>>>(pj, w_conv, b_conv, uf, ubf);

    // 5) a = sigmoid(u@w_a^T+b_a), b = sigmoid(...), c = tanh(...)
    {
        dim3 grid(INNERc / 256, (int)(Mrows / 128));
        wmma_gemm_kernel<MODE_SIG><<<grid, 256, 0, stream>>>(ubf, wAB, b_a, aAct,
                                                             INNERc, INNERc, nullptr, nullptr);
        wmma_gemm_kernel<MODE_SIG><<<grid, 256, 0, stream>>>(ubf, wBB, b_b, bAct,
                                                             INNERc, INNERc, nullptr, nullptr);
        wmma_gemm_kernel<MODE_TANH><<<grid, 256, 0, stream>>>(ubf, wCB, b_c, cAct,
                                                              INNERc, INNERc, nullptr, nullptr);
    }

    // 6) sequential scan -> y (bf16)
    scan_kernel<<<(Bb * INNERc + 255) / 256, 256, 0, stream>>>(aAct, bAct, cAct, uf, mask, ybf);

    // 7) out = x + y @ w_out^T + b_out
    {
        dim3 grid(Dd / 256, (int)(Mrows / 128));
        wmma_gemm_kernel<MODE_OUT><<<grid, 256, 0, stream>>>(ybf, wOutB, b_out, outp,
                                                             Dd, INNERc, nullptr, x);
    }
    (void)in_sizes; (void)n_in; (void)out_size; (void)ws_size;
}
